// STGEN_28552942584334
// MI455X (gfx1250) — matrix operand order repaired and checked by reading, not compiled, not run on hardware
//
#include <hip/hip_runtime.h>
#include <hip/hip_bf16.h>

#define N_NODES 100000
#define N_EDGES 1600000
#define IN_DIM  128
#define BN_EPS  1e-5f

typedef _Float16 half16 __attribute__((ext_vector_type(16)));
typedef _Float16 half8  __attribute__((ext_vector_type(8)));
typedef float    f32x8  __attribute__((ext_vector_type(8)));

static __device__ __forceinline__ f32x8 wmma_f16(half16 a, half16 b, f32x8 c) {
  return __builtin_amdgcn_wmma_f32_16x16x32_f16(false, a, false, b, (short)0, c, false, false);
}
static __device__ __forceinline__ float leaky(float v) { return v >= 0.f ? v : 0.01f * v; }

// D-layout (two 16x16 tiles, 32 out dims) -> B-layout (K=32) for the next WMMA.
// lane<16 holds col n rows v / v+8 of tile0 after swap; lane>=16 gets tile1.
static __device__ __forceinline__ half16 chain_B32(const float e0[8], const float e1[8], int hi) {
  float q0[8], q1[8];
#pragma unroll
  for (int v = 0; v < 8; ++v) { q0[v] = __shfl_xor(e0[v], 16, 32); q1[v] = __shfl_xor(e1[v], 16, 32); }
  half16 b;
#pragma unroll
  for (int v = 0; v < 8; ++v) {
    b[v]     = hi ? (_Float16)q1[v] : (_Float16)e0[v];
    b[v + 8] = hi ? (_Float16)e1[v] : (_Float16)q0[v];
  }
  return b;
}
// Single 16-dim tile -> B-layout with K=16..31 zero-padded.
static __device__ __forceinline__ half16 chain_B16(const float e[8], int hi) {
  float q[8];
#pragma unroll
  for (int v = 0; v < 8; ++v) q[v] = __shfl_xor(e[v], 16, 32);
  half16 b;
#pragma unroll
  for (int v = 0; v < 8; ++v) {
    b[v]     = hi ? (_Float16)0.f : (_Float16)e[v];
    b[v + 8] = hi ? (_Float16)0.f : (_Float16)q[v];
  }
  return b;
}

// Fused edge encoder: ea = leaky(edge_attr@W1+b1)@W2+b2 for 16 edges per wave.
// Output: ea[v] = dim (v+8*hi) of this lane's edge; oea[v] = partner's ea[v].
static __device__ __forceinline__ void edge_encode(const float* ep, int hi,
    half16 aee1, half16 aee2, const float* eb1, const float* eb2,
    float ea[8], float oea[8]) {
  half16 bea;
  float epv[16];
#pragma unroll
  for (int j = 0; j < 16; ++j) epv[j] = ep[j];   // hi lanes read same line as partner
#pragma unroll
  for (int j = 0; j < 16; ++j) bea[j] = hi ? (_Float16)0.f : (_Float16)epv[j];
  f32x8 z = {0.f,0.f,0.f,0.f,0.f,0.f,0.f,0.f};
  f32x8 t1 = wmma_f16(aee1, bea, z);
  float e0[8];
#pragma unroll
  for (int v = 0; v < 8; ++v) e0[v] = leaky(t1[v] + eb1[v + 8 * hi]);
  half16 bm = chain_B16(e0, hi);
  f32x8 t2 = wmma_f16(aee2, bm, z);
#pragma unroll
  for (int v = 0; v < 8; ++v) ea[v] = t2[v] + eb2[v + 8 * hi];
#pragma unroll
  for (int v = 0; v < 8; ++v) oea[v] = __shfl_xor(ea[v], 16, 32);
}

// ---------------- weight pre-pack into WMMA A-layout (17 tiles of 512 halves) ---
// A-layout per lane: half j -> K = j + (hi?16:0), row M = lane&15 (+mbase)
__global__ void prep_weights(_Float16* __restrict__ Apk,
    const float* ne_w1, const float* ne_w2, const float* ee_w1, const float* ee_w2,
    const float* nn1_w1, const float* nn1_w2, const float* nn1_w3,
    const float* nn2_w1, const float* nn2_w2, const float* nn2_w3) {
  int lane = threadIdx.x;
  if (lane >= 32) return;
  int hi = lane >> 4, mloc = lane & 15;
  auto pack = [&](int tile, const float* W, int ncols, int mbase, int klim, int rofs, int mode) {
#pragma unroll
    for (int j = 0; j < 16; ++j) {
      int k = j + (hi ? 16 : 0);
      int row;
      if (mode == 1) row = (k < 8) ? k : (k < 16 ? -1 : k - 8);  // conv2 L1: [x1(8) | pad(8) | ea(16)]
      else           row = (k < klim) ? (k + rofs) : -1;
      int m = mbase + mloc;
      float v = (row >= 0 && m < ncols) ? W[row * ncols + m] : 0.f;
      Apk[tile * 512 + lane * 16 + j] = (_Float16)v;
    }
  };
  pack(0,  ne_w1, 16, 0, 32, 0,  0);   // node enc W1, K-chunks 0..3
  pack(1,  ne_w1, 16, 0, 32, 32, 0);
  pack(2,  ne_w1, 16, 0, 32, 64, 0);
  pack(3,  ne_w1, 16, 0, 32, 96, 0);
  pack(4,  ne_w2, 16, 0, 16, 0,  0);
  pack(5,  ee_w1, 16, 0, 16, 0,  0);
  pack(6,  ee_w2, 16, 0, 16, 0,  0);
  pack(7,  nn1_w1, 32, 0,  32, 0, 0);
  pack(8,  nn1_w1, 32, 16, 32, 0, 0);
  pack(9,  nn1_w2, 32, 0,  32, 0, 0);
  pack(10, nn1_w2, 32, 16, 32, 0, 0);
  pack(11, nn1_w3, 8,  0,  32, 0, 0);
  pack(12, nn2_w1, 24, 0,  0, 0, 1);
  pack(13, nn2_w1, 24, 16, 0, 0, 1);
  pack(14, nn2_w2, 24, 0,  24, 0, 0);
  pack(15, nn2_w2, 24, 16, 24, 0, 0);
  pack(16, nn2_w3, 10, 0,  24, 0, 0);
}

__global__ void zero_f32(float* __restrict__ p, int n) {
  int i = blockIdx.x * blockDim.x + threadIdx.x;
  if (i < n) p[i] = 0.f;
}

// ---------------- node encoder: h = (leaky(x@W1+b1)@W2+b2) * (t*tw_w+tw_b) -------
__global__ __launch_bounds__(256) void node_encoder(
    const float* __restrict__ x, const float* __restrict__ tw_in,
    const float* __restrict__ ne_b1, const float* __restrict__ ne_b2,
    const float* __restrict__ tw_w, const float* __restrict__ tw_b,
    const _Float16* __restrict__ Apk,
    float* __restrict__ h_f32, _Float16* __restrict__ h_h) {
  const int lane = threadIdx.x & 31, hi = lane >> 4, n = lane & 15;
  const int wpb = blockDim.x >> 5;
  const int nwaves = gridDim.x * wpb;
  half16 aq[4], a2;
#pragma unroll
  for (int q = 0; q < 4; ++q) aq[q] = *(const half16*)(Apk + (q * 32 + lane) * 16);
  a2 = *(const half16*)(Apk + (4 * 32 + lane) * 16);
  f32x8 z = {0.f,0.f,0.f,0.f,0.f,0.f,0.f,0.f};
  for (int t = blockIdx.x * wpb + (threadIdx.x >> 5); t < N_NODES / 16; t += nwaves) {
    int node = t * 16 + n;
    f32x8 c = z;
#pragma unroll
    for (int q = 0; q < 4; ++q) {
      const float* bp = x + (size_t)node * IN_DIM + q * 32 + (hi ? 16 : 0);
      half16 b;
#pragma unroll
      for (int j = 0; j < 16; ++j) b[j] = (_Float16)bp[j];
      c = wmma_f16(aq[q], b, c);
    }
    float e0[8];
#pragma unroll
    for (int v = 0; v < 8; ++v) e0[v] = leaky(c[v] + ne_b1[v + 8 * hi]);
    half16 b2 = chain_B16(e0, hi);
    f32x8 hd = wmma_f16(a2, b2, z);
    float tv = tw_in[node];
    float hv[8];
#pragma unroll
    for (int v = 0; v < 8; ++v) {
      int m = v + 8 * hi;
      hv[v] = (hd[v] + ne_b2[m]) * (tv * tw_w[m] + tw_b[m]);
    }
    float* hp = h_f32 + (size_t)node * 16 + 8 * hi;
    float4 f0; f0.x = hv[0]; f0.y = hv[1]; f0.z = hv[2]; f0.w = hv[3];
    float4 f1; f1.x = hv[4]; f1.y = hv[5]; f1.z = hv[6]; f1.w = hv[7];
    *(float4*)(hp) = f0; *(float4*)(hp + 4) = f1;
    half8 hph;
#pragma unroll
    for (int v = 0; v < 8; ++v) hph[v] = (_Float16)hv[v];
    *(half8*)(h_h + (size_t)node * 16 + 8 * hi) = hph;
  }
}

// ---------------- conv1 edge pipeline: encode ea, MLP 32->32->32->8, scatter ----
__global__ __launch_bounds__(256) void conv1_edges(
    const float* __restrict__ eattr, const int* __restrict__ src, const int* __restrict__ dst,
    const _Float16* __restrict__ h_h, const _Float16* __restrict__ Apk,
    const float* __restrict__ eb1, const float* __restrict__ eb2,
    const float* __restrict__ b1, const float* __restrict__ b2, const float* __restrict__ b3,
    float* __restrict__ agg1) {
  const int lane = threadIdx.x & 31, hi = lane >> 4, n = lane & 15;
  const int wpb = blockDim.x >> 5;
  const int nwaves = gridDim.x * wpb;
  half16 aee1 = *(const half16*)(Apk + (5 * 32 + lane) * 16);
  half16 aee2 = *(const half16*)(Apk + (6 * 32 + lane) * 16);
  half16 a10  = *(const half16*)(Apk + (7 * 32 + lane) * 16);
  half16 a11  = *(const half16*)(Apk + (8 * 32 + lane) * 16);
  half16 a20  = *(const half16*)(Apk + (9 * 32 + lane) * 16);
  half16 a21  = *(const half16*)(Apk + (10 * 32 + lane) * 16);
  half16 a3   = *(const half16*)(Apk + (11 * 32 + lane) * 16);
  f32x8 z = {0.f,0.f,0.f,0.f,0.f,0.f,0.f,0.f};
  for (int t = blockIdx.x * wpb + (threadIdx.x >> 5); t < N_EDGES / 16; t += nwaves) {
    int e = t * 16 + n;
    int s = src[e], dnode = dst[e];
    float ea[8], oea[8];
    edge_encode(eattr + (size_t)e * 16, hi, aee1, aee2, eb1, eb2, ea, oea);
    // B0: low lanes K=0..15 = h[src]; hi lanes K=16..31 = ea dims 0..15
    half16 hload = *(const half16*)(h_h + (size_t)s * 16);
    half16 b0;
#pragma unroll
    for (int v = 0; v < 8; ++v) {
      b0[v]     = hi ? (_Float16)oea[v] : hload[v];
      b0[v + 8] = hi ? (_Float16)ea[v]  : hload[v + 8];
    }
    f32x8 d0 = wmma_f16(a10, b0, z), d1 = wmma_f16(a11, b0, z);
    float e0[8], e1[8];
#pragma unroll
    for (int v = 0; v < 8; ++v) {
      e0[v] = leaky(d0[v] + b1[v + 8 * hi]);
      e1[v] = leaky(d1[v] + b1[16 + v + 8 * hi]);
    }
    half16 bl2 = chain_B32(e0, e1, hi);
    d0 = wmma_f16(a20, bl2, z); d1 = wmma_f16(a21, bl2, z);
#pragma unroll
    for (int v = 0; v < 8; ++v) {
      e0[v] = leaky(d0[v] + b2[v + 8 * hi]);
      e1[v] = leaky(d1[v] + b2[16 + v + 8 * hi]);
    }
    half16 bl3 = chain_B32(e0, e1, hi);
    f32x8 d3 = wmma_f16(a3, bl3, z);
    float f[8];
#pragma unroll
    for (int v = 0; v < 8; ++v) f[v] = d3[v] + b3[v];   // valid on low lanes (m=0..7)
    float g[4];
#pragma unroll
    for (int i = 0; i < 4; ++i) g[i] = __shfl_xor(f[4 + i], 16, 32);
    float* ap = agg1 + (size_t)dnode * 8;
#pragma unroll
    for (int i = 0; i < 4; ++i) {
      int m = hi ? 4 + i : i;
      float val = hi ? g[i] : f[i];
      atomicAdd(ap + m, val);
    }
  }
}

// ---------------- conv1 node finalize + BN batch-stats ---------------------------
__global__ __launch_bounds__(256) void conv1_nodes(
    const float* __restrict__ agg1, const float* __restrict__ h_f32,
    const float* __restrict__ root1, const float* __restrict__ bias1,
    float* __restrict__ x1, float* __restrict__ bn) {
  __shared__ float ssum[8], ssq[8];
  int tid = threadIdx.x;
  if (tid < 8) { ssum[tid] = 0.f; ssq[tid] = 0.f; }
  __syncthreads();
  int idx = blockIdx.x * blockDim.x + tid;
  if (idx < N_NODES) {
    float hv[16];
#pragma unroll
    for (int k = 0; k < 16; ++k) hv[k] = h_f32[(size_t)idx * 16 + k];
#pragma unroll
    for (int j = 0; j < 8; ++j) {
      float r = agg1[(size_t)idx * 8 + j] + bias1[j];
#pragma unroll
      for (int k = 0; k < 16; ++k) r += hv[k] * root1[k * 8 + j];
      r = fmaxf(r, 0.f);
      x1[(size_t)idx * 8 + j] = r;
      atomicAdd(&ssum[j], r);
      atomicAdd(&ssq[j], r * r);
    }
  }
  __syncthreads();
  if (tid < 8) { atomicAdd(&bn[tid], ssum[tid]); atomicAdd(&bn[8 + tid], ssq[tid]); }
}

__global__ void bn_finalize(float* __restrict__ bn, const float* __restrict__ g, const float* __restrict__ b) {
  int j = threadIdx.x;
  if (j < 8) {
    float mu = bn[j] / (float)N_NODES;
    float var = bn[8 + j] / (float)N_NODES - mu * mu;
    float sc = g[j] * rsqrtf(var + BN_EPS);
    bn[16 + j] = sc;
    bn[24 + j] = b[j] - mu * sc;
  }
}

__global__ __launch_bounds__(256) void bn_apply(
    const float* __restrict__ x1, const float* __restrict__ bn,
    float* __restrict__ x1n, _Float16* __restrict__ x1n_h) {
  int idx = blockIdx.x * blockDim.x + threadIdx.x;
  if (idx >= N_NODES) return;
  half8 hx;
#pragma unroll
  for (int j = 0; j < 8; ++j) {
    float v = x1[(size_t)idx * 8 + j] * bn[16 + j] + bn[24 + j];
    x1n[(size_t)idx * 8 + j] = v;
    hx[j] = (_Float16)v;
  }
  *(half8*)(x1n_h + (size_t)idx * 8) = hx;
}

// ---------------- conv2 edge pipeline: MLP 24->24->24->10 (K padded to 32) -------
__global__ __launch_bounds__(256) void conv2_edges(
    const float* __restrict__ eattr, const int* __restrict__ src, const int* __restrict__ dst,
    const _Float16* __restrict__ x1n_h, const _Float16* __restrict__ Apk,
    const float* __restrict__ eb1, const float* __restrict__ eb2,
    const float* __restrict__ b1, const float* __restrict__ b2, const float* __restrict__ b3,
    float* __restrict__ agg2) {
  const int lane = threadIdx.x & 31, hi = lane >> 4, n = lane & 15;
  const int wpb = blockDim.x >> 5;
  const int nwaves = gridDim.x * wpb;
  half16 aee1 = *(const half16*)(Apk + (5 * 32 + lane) * 16);
  half16 aee2 = *(const half16*)(Apk + (6 * 32 + lane) * 16);
  half16 a10  = *(const half16*)(Apk + (12 * 32 + lane) * 16);
  half16 a11  = *(const half16*)(Apk + (13 * 32 + lane) * 16);
  half16 a20  = *(const half16*)(Apk + (14 * 32 + lane) * 16);
  half16 a21  = *(const half16*)(Apk + (15 * 32 + lane) * 16);
  half16 a3   = *(const half16*)(Apk + (16 * 32 + lane) * 16);
  f32x8 z = {0.f,0.f,0.f,0.f,0.f,0.f,0.f,0.f};
  for (int t = blockIdx.x * wpb + (threadIdx.x >> 5); t < N_EDGES / 16; t += nwaves) {
    int e = t * 16 + n;
    int s = src[e], dnode = dst[e];
    float ea[8], oea[8];
    edge_encode(eattr + (size_t)e * 16, hi, aee1, aee2, eb1, eb2, ea, oea);
    // B0: low lanes K=0..7 = x1n[src], K=8..15 = 0; hi lanes K=16..31 = ea dims 0..15
    half8 xl = *(const half8*)(x1n_h + (size_t)s * 8);
    half16 b0;
#pragma unroll
    for (int v = 0; v < 8; ++v) {
      b0[v]     = hi ? (_Float16)oea[v] : xl[v];
      b0[v + 8] = hi ? (_Float16)ea[v]  : (_Float16)0.f;
    }
    f32x8 d0 = wmma_f16(a10, b0, z), d1 = wmma_f16(a11, b0, z);
    float e0[8], e1[8];
#pragma unroll
    for (int v = 0; v < 8; ++v) {
      int i2 = 16 + v + 8 * hi;
      e0[v] = leaky(d0[v] + b1[v + 8 * hi]);
      e1[v] = leaky(d1[v] + (i2 < 24 ? b1[i2] : 0.f));   // rows 24..31 are zero-pad
    }
    half16 bl2 = chain_B32(e0, e1, hi);
    d0 = wmma_f16(a20, bl2, z); d1 = wmma_f16(a21, bl2, z);
#pragma unroll
    for (int v = 0; v < 8; ++v) {
      int i2 = 16 + v + 8 * hi;
      e0[v] = leaky(d0[v] + b2[v + 8 * hi]);
      e1[v] = leaky(d1[v] + (i2 < 24 ? b2[i2] : 0.f));
    }
    half16 bl3 = chain_B32(e0, e1, hi);
    f32x8 d3 = wmma_f16(a3, bl3, z);
    float f[8];
#pragma unroll
    for (int v = 0; v < 8; ++v) f[v] = d3[v] + b3[v];   // valid for low lanes m=0..7
    float fh0 = d3[0] + b3[8], fh1 = d3[1] + b3[9];     // hi lanes own m=8,9
    float o5 = __shfl_xor(f[5], 16, 32);
    float o6 = __shfl_xor(f[6], 16, 32);
    float o7 = __shfl_xor(f[7], 16, 32);
    float* ap = agg2 + (size_t)dnode * 10;
    // balanced 5-atomic split: low does m=0..4, hi does m={8,9,5,6,7}
    atomicAdd(ap + (hi ? 8 : 0), hi ? fh0 : f[0]);
    atomicAdd(ap + (hi ? 9 : 1), hi ? fh1 : f[1]);
    atomicAdd(ap + (hi ? 5 : 2), hi ? o5 : f[2]);
    atomicAdd(ap + (hi ? 6 : 3), hi ? o6 : f[3]);
    atomicAdd(ap + (hi ? 7 : 4), hi ? o7 : f[4]);
  }
}

// ---------------- final: x2, fc, log_softmax ------------------------------------
__global__ __launch_bounds__(256) void final_nodes(
    const float* __restrict__ agg2, const float* __restrict__ x1n,
    const float* __restrict__ root2, const float* __restrict__ bias2,
    const float* __restrict__ fc_w, const float* __restrict__ fc_b,
    float* __restrict__ out) {
  int idx = blockIdx.x * blockDim.x + threadIdx.x;
  if (idx >= N_NODES) return;
  float a[8];
#pragma unroll
  for (int k = 0; k < 8; ++k) a[k] = x1n[(size_t)idx * 8 + k];
  float x2[10];
#pragma unroll
  for (int c = 0; c < 10; ++c) {
    float r = agg2[(size_t)idx * 10 + c] + bias2[c];
#pragma unroll
    for (int k = 0; k < 8; ++k) r += a[k] * root2[k * 10 + c];
    x2[c] = fmaxf(r, 0.f);
  }
  float o[10], mx = -3.4e38f;
#pragma unroll
  for (int c = 0; c < 10; ++c) {
    float s = fc_b[c];
#pragma unroll
    for (int k = 0; k < 8; ++k) s += a[k] * fc_w[k * 10 + c];
#pragma unroll
    for (int k = 0; k < 10; ++k) s += x2[k] * fc_w[(8 + k) * 10 + c];
    o[c] = s;
    mx = fmaxf(mx, s);
  }
  float se = 0.f;
#pragma unroll
  for (int c = 0; c < 10; ++c) se += __expf(o[c] - mx);
  float lse = __logf(se);
#pragma unroll
  for (int c = 0; c < 10; ++c) out[(size_t)idx * 10 + c] = o[c] - mx - lse;
}

extern "C" void kernel_launch(void* const* d_in, const int* in_sizes, int n_in,
                              void* d_out, int out_size, void* d_ws, size_t ws_size,
                              hipStream_t stream) {
  (void)in_sizes; (void)n_in; (void)out_size; (void)ws_size;
  const float* x     = (const float*)d_in[0];
  const float* eattr = (const float*)d_in[1];
  const float* tw_in = (const float*)d_in[2];
  const int*   eidx  = (const int*)d_in[3];
  const float *ne_w1 = (const float*)d_in[4],  *ne_b1 = (const float*)d_in[5];
  const float *ne_w2 = (const float*)d_in[6],  *ne_b2 = (const float*)d_in[7];
  const float *ee_w1 = (const float*)d_in[8],  *ee_b1 = (const float*)d_in[9];
  const float *ee_w2 = (const float*)d_in[10], *ee_b2 = (const float*)d_in[11];
  const float *tw_w  = (const float*)d_in[12], *tw_b  = (const float*)d_in[13];
  const float *nn1_w1 = (const float*)d_in[14], *nn1_b1 = (const float*)d_in[15];
  const float *nn1_w2 = (const float*)d_in[16], *nn1_b2 = (const float*)d_in[17];
  const float *nn1_w3 = (const float*)d_in[18], *nn1_b3 = (const float*)d_in[19];
  const float *root1  = (const float*)d_in[20], *bias1  = (const float*)d_in[21];
  const float *bn_g   = (const float*)d_in[22], *bn_b   = (const float*)d_in[23];
  const float *nn2_w1 = (const float*)d_in[24], *nn2_b1 = (const float*)d_in[25];
  const float *nn2_w2 = (const float*)d_in[26], *nn2_b2 = (const float*)d_in[27];
  const float *nn2_w3 = (const float*)d_in[28], *nn2_b3 = (const float*)d_in[29];
  const float *root2  = (const float*)d_in[30], *bias2  = (const float*)d_in[31];
  const float *fc_w   = (const float*)d_in[32], *fc_b   = (const float*)d_in[33];
  const int* src = eidx;
  const int* dst = eidx + N_EDGES;
  float* out = (float*)d_out;

  char* base = (char*)d_ws;
  size_t off = 0;
  auto carve = [&](size_t bytes) -> char* {
    char* p = base + off;
    off = (off + bytes + 255) & ~(size_t)255;
    return p;
  };
  _Float16* Apk   = (_Float16*)carve(17 * 512 * 2);
  float*    h_f32 = (float*)carve((size_t)N_NODES * 16 * 4);
  _Float16* h_h   = (_Float16*)carve((size_t)N_NODES * 16 * 2);
  float*    x1    = (float*)carve((size_t)N_NODES * 8 * 4);
  float*    x1n   = (float*)carve((size_t)N_NODES * 8 * 4);
  _Float16* x1n_h = (_Float16*)carve((size_t)N_NODES * 8 * 2);
  float*    agg1  = (float*)carve((size_t)N_NODES * 8 * 4);
  float*    agg2  = (float*)carve((size_t)N_NODES * 10 * 4);
  float*    bn    = (float*)carve(32 * 4);

  prep_weights<<<1, 32, 0, stream>>>(Apk, ne_w1, ne_w2, ee_w1, ee_w2,
                                     nn1_w1, nn1_w2, nn1_w3, nn2_w1, nn2_w2, nn2_w3);
  zero_f32<<<(N_NODES * 8 + 255) / 256, 256, 0, stream>>>(agg1, N_NODES * 8);
  zero_f32<<<(N_NODES * 10 + 255) / 256, 256, 0, stream>>>(agg2, N_NODES * 10);
  zero_f32<<<1, 32, 0, stream>>>(bn, 32);
  node_encoder<<<512, 256, 0, stream>>>(x, tw_in, ne_b1, ne_b2, tw_w, tw_b, Apk, h_f32, h_h);
  conv1_edges<<<1280, 256, 0, stream>>>(eattr, src, dst, h_h, Apk, ee_b1, ee_b2,
                                        nn1_b1, nn1_b2, nn1_b3, agg1);
  conv1_nodes<<<(N_NODES + 255) / 256, 256, 0, stream>>>(agg1, h_f32, root1, bias1, x1, bn);
  bn_finalize<<<1, 32, 0, stream>>>(bn, bn_g, bn_b);
  bn_apply<<<(N_NODES + 255) / 256, 256, 0, stream>>>(x1, bn, x1n, x1n_h);
  conv2_edges<<<1280, 256, 0, stream>>>(eattr, src, dst, x1n_h, Apk, ee_b1, ee_b2,
                                        nn2_b1, nn2_b2, nn2_b3, agg2);
  final_nodes<<<(N_NODES + 255) / 256, 256, 0, stream>>>(agg2, x1n, root2, bias2, fc_w, fc_b, out);
}
